// OctreeAttention_68204080660877
// MI455X (gfx1250) — matrix operand order repaired and checked by reading, not compiled, not run on hardware
//
#include <hip/hip_runtime.h>
#include <hip/hip_bf16.h>

// Problem constants (match the reference)
#define NW_  8192
#define KW   32     // window size
#define CD   256    // dim
#define HH   8      // heads
#define DD   32     // head dim
#define POSB 25
#define RPEN 51

typedef __attribute__((ext_vector_type(16))) __bf16 v16bf;
typedef __attribute__((ext_vector_type(8)))  float  v8f;

union BF16Frag {
  v16bf  v;
  __bf16 e[16];
  uint4  q[2];
};

__device__ __forceinline__ v8f wmma_bf16(v16bf a, v16bf b, v8f c) {
  // D = A(16x32 bf16) x B(32x16 bf16) + C(16x16 f32)
  return __builtin_amdgcn_wmma_f32_16x16x32_bf16(
      /*neg_a=*/false, a, /*neg_b=*/false, b,
      /*c_mod=*/(short)0, c, /*reuse_a=*/false, /*reuse_b=*/false);
}

// ---------------------------------------------------------------------------
// Weight pre-transpose + fp32->bf16:  WT[n][k] = (bf16) W[k][n]
// ---------------------------------------------------------------------------
__global__ void wtrans_kernel(const float* __restrict__ W, __bf16* __restrict__ WT,
                              int Kdim, int Ndim) {
  int idx = blockIdx.x * blockDim.x + threadIdx.x;
  if (idx >= Kdim * Ndim) return;
  int k = idx / Ndim, n = idx % Ndim;     // coalesced read of W row-major
  WT[(size_t)n * Kdim + k] = (__bf16)W[idx];
}

// ---------------------------------------------------------------------------
// Kernel 1: QKV GEMM.  Y(262144 x 768) = data(262144 x 256) @ W_qkv + b_qkv
// One wave computes a 32x32 output tile via 2x2 WMMA accumulators.
// Grid: x = n-tiles (fastest, 24) so adjacent blocks reuse the same 32 A-rows
// out of L2; y = m-tiles (8192). A is then read from HBM exactly once.
// Epilogue scatters into per-head layouts:
//   q: (NW,H,K,d) bf16, pre-scaled by d^-0.5
//   k: (NW,H,K,d) bf16
//   v: (NW,H,d,K) bf16  (TRANSPOSED so attn@v B-frags are contiguous rows)
// ---------------------------------------------------------------------------
__global__ __launch_bounds__(32) void qkv_gemm_kernel(
    const float* __restrict__ data, const __bf16* __restrict__ WT,
    const float* __restrict__ b_qkv,
    __bf16* __restrict__ qws, __bf16* __restrict__ kws, __bf16* __restrict__ vtws)
{
  const int lane = threadIdx.x;
  const int mL = lane & 15, hi = lane >> 4;
  const int n0 = blockIdx.x * 32;   // n fastest -> L2 reuse of A rows
  const int m0 = blockIdx.y * 32;

  v8f acc00 = {}, acc01 = {}, acc10 = {}, acc11 = {};

  for (int kk = 0; kk < CD; kk += 32) {
    BF16Frag a[2], b[2];
#pragma unroll
    for (int t = 0; t < 2; ++t) {
      // A-frag: row (m0+t*16+mL), two 8-elem chunks at k = 8*hi and 16+8*hi
      const float* ap = data + (size_t)(m0 + t * 16 + mL) * CD + kk;
      const float4 f0 = *(const float4*)(ap + 8 * hi);
      const float4 f1 = *(const float4*)(ap + 8 * hi + 4);
      const float4 f2 = *(const float4*)(ap + 16 + 8 * hi);
      const float4 f3 = *(const float4*)(ap + 16 + 8 * hi + 4);
      a[t].e[0]  = (__bf16)f0.x; a[t].e[1]  = (__bf16)f0.y;
      a[t].e[2]  = (__bf16)f0.z; a[t].e[3]  = (__bf16)f0.w;
      a[t].e[4]  = (__bf16)f1.x; a[t].e[5]  = (__bf16)f1.y;
      a[t].e[6]  = (__bf16)f1.z; a[t].e[7]  = (__bf16)f1.w;
      a[t].e[8]  = (__bf16)f2.x; a[t].e[9]  = (__bf16)f2.y;
      a[t].e[10] = (__bf16)f2.z; a[t].e[11] = (__bf16)f2.w;
      a[t].e[12] = (__bf16)f3.x; a[t].e[13] = (__bf16)f3.y;
      a[t].e[14] = (__bf16)f3.z; a[t].e[15] = (__bf16)f3.w;
      // B-frag: WT row n, two 8-elem chunks at k = 8*hi and 16+8*hi
      const __bf16* bp = WT + (size_t)(n0 + t * 16 + mL) * CD + kk;
      b[t].q[0] = *(const uint4*)(bp + 8 * hi);
      b[t].q[1] = *(const uint4*)(bp + 16 + 8 * hi);
    }
    acc00 = wmma_bf16(a[0].v, b[0].v, acc00);
    acc01 = wmma_bf16(a[0].v, b[1].v, acc01);
    acc10 = wmma_bf16(a[1].v, b[0].v, acc10);
    acc11 = wmma_bf16(a[1].v, b[1].v, acc11);
  }

  const float scale = 0.17677669529663687f;  // 32^-0.5
  v8f accs[4] = {acc00, acc01, acc10, acc11};
#pragma unroll
  for (int mt = 0; mt < 2; ++mt) {
#pragma unroll
    for (int nt = 0; nt < 2; ++nt) {
      const int n = n0 + nt * 16 + mL;
      const int which = n >> 8;        // 0=q 1=k 2=v (uniform per n-tile)
      const int h  = (n >> 5) & 7;
      const int dd = n & 31;
      const float bias = b_qkv[n];
      if (which == 2) {
        // v: lane writes 8 consecutive positions i of row dd -> one b128 store
        union { __bf16 e[8]; uint4 q; } pack;
#pragma unroll
        for (int r = 0; r < 8; ++r)
          pack.e[r] = (__bf16)(accs[mt * 2 + nt][r] + bias);
        const int mb = m0 + mt * 16 + 8 * hi;   // 8-aligned, stays in one window
        const int nw = mb >> 5, ib = mb & 31;
        *(uint4*)&vtws[((size_t)(nw * HH + h) * DD + dd) * KW + ib] = pack.q;
      } else {
#pragma unroll
        for (int r = 0; r < 8; ++r) {
          const int m  = m0 + mt * 16 + r + 8 * hi;  // C/D layout: M = r + 8*hi
          const int nw = m >> 5, i = m & 31;
          const float val = accs[mt * 2 + nt][r] + bias;
          if (which == 0)
            qws[((size_t)(nw * HH + h) * KW + i) * DD + dd] = (__bf16)(val * scale);
          else
            kws[((size_t)(nw * HH + h) * KW + i) * DD + dd] = (__bf16)val;
        }
      }
    }
  }
}

// ---------------------------------------------------------------------------
// Kernel 2: per-(window, head) attention. One wave per (nw, h).
// scores = q @ k^T (4 WMMAs) -> LDS -> +RPE +mask, fp32 softmax per row
// -> bf16 probs in LDS -> out = P @ v (4 WMMAs, B = v^T rows) -> bf16 out.
// rel_pos / mask rows are 16B-aligned: load as int4 / float4 (b128).
// ---------------------------------------------------------------------------
__global__ __launch_bounds__(32) void attn_kernel(
    const __bf16* __restrict__ qws, const __bf16* __restrict__ kws,
    const __bf16* __restrict__ vtws,
    const float* __restrict__ mask, const int* __restrict__ rel_pos,
    const float* __restrict__ rpe_table,
    __bf16* __restrict__ attnout)
{
  __shared__ __align__(16) float  sS[KW * KW];
  __shared__ __align__(16) __bf16 sP[KW * KW];

  const int nw = blockIdx.x;
  const int h  = blockIdx.y;
  const int lane = threadIdx.x;
  const int mL = lane & 15, hi = lane >> 4;

  const __bf16* qb = qws  + (size_t)(nw * HH + h) * (KW * DD);
  const __bf16* kb = kws  + (size_t)(nw * HH + h) * (KW * DD);
  const __bf16* vb = vtws + (size_t)(nw * HH + h) * (DD * KW);

  // ---- scores = q @ k^T : A = q rows, B^T = k rows (both contiguous) ----
  BF16Frag aq[2], bk[2];
#pragma unroll
  for (int t = 0; t < 2; ++t) {
    const __bf16* ap = qb + (size_t)(t * 16 + mL) * DD;
    aq[t].q[0] = *(const uint4*)(ap + 8 * hi);
    aq[t].q[1] = *(const uint4*)(ap + 16 + 8 * hi);
    const __bf16* bp = kb + (size_t)(t * 16 + mL) * DD;
    bk[t].q[0] = *(const uint4*)(bp + 8 * hi);
    bk[t].q[1] = *(const uint4*)(bp + 16 + 8 * hi);
  }
  v8f s00 = {}, s01 = {}, s10 = {}, s11 = {};
  s00 = wmma_bf16(aq[0].v, bk[0].v, s00);
  s01 = wmma_bf16(aq[0].v, bk[1].v, s01);
  s10 = wmma_bf16(aq[1].v, bk[0].v, s10);
  s11 = wmma_bf16(aq[1].v, bk[1].v, s11);

  v8f ss[4] = {s00, s01, s10, s11};
#pragma unroll
  for (int mt = 0; mt < 2; ++mt)
#pragma unroll
    for (int nt = 0; nt < 2; ++nt)
#pragma unroll
      for (int r = 0; r < 8; ++r)
        sS[(mt * 16 + r + 8 * hi) * KW + nt * 16 + mL] = ss[mt * 2 + nt][r];
  __syncthreads();

  // ---- per-row RPE + mask + softmax (lane i owns row i) ----
  const int i = lane;
  const float4* mrow4 = (const float4*)(mask + ((size_t)nw * KW + i) * KW);
  const int4*   rrow4 = (const int4*)(rel_pos + (((size_t)nw * KW + i) * KW) * 3);
  float buf[KW];
  float mx = -3.0e38f;
#pragma unroll
  for (int jb = 0; jb < KW / 4; ++jb) {
    const float4 mk = mrow4[jb];
    const int4 ra = rrow4[jb * 3 + 0];
    const int4 rb = rrow4[jb * 3 + 1];
    const int4 rc = rrow4[jb * 3 + 2];
    const int xs[4] = {ra.x, ra.w, rb.z, rc.y};
    const int ys[4] = {ra.y, rb.x, rb.w, rc.z};
    const int zs[4] = {ra.z, rb.y, rc.x, rc.w};
    const float mkv[4] = {mk.x, mk.y, mk.z, mk.w};
#pragma unroll
    for (int u = 0; u < 4; ++u) {
      const int j = jb * 4 + u;
      const int x = min(max(xs[u], -POSB), POSB);
      const int y = min(max(ys[u], -POSB), POSB);
      const int z = min(max(zs[u], -POSB), POSB);
      const float bias = rpe_table[(x + POSB) * HH + h]
                       + rpe_table[(RPEN + y + POSB) * HH + h]
                       + rpe_table[(2 * RPEN + z + POSB) * HH + h];
      const float v = sS[i * KW + j] + bias + mkv[u];
      buf[j] = v;
      mx = fmaxf(mx, v);
    }
  }
  float sum = 0.f;
#pragma unroll 4
  for (int j = 0; j < KW; ++j) {
    const float e = __expf(buf[j] - mx);
    buf[j] = e;
    sum += e;
  }
  const float inv = 1.0f / sum;
#pragma unroll 4
  for (int j = 0; j < KW; ++j)
    sP[i * KW + j] = (__bf16)(buf[j] * inv);
  __syncthreads();

  // ---- out = P @ v : A = P rows (LDS), B^T = v^T rows (contiguous) ----
  BF16Frag ap2[2], bv[2];
#pragma unroll
  for (int t = 0; t < 2; ++t) {
    const __bf16* ap = sP + (t * 16 + mL) * KW;
    ap2[t].q[0] = *(const uint4*)(ap + 8 * hi);
    ap2[t].q[1] = *(const uint4*)(ap + 16 + 8 * hi);
    const __bf16* bp = vb + (size_t)(t * 16 + mL) * KW;
    bv[t].q[0] = *(const uint4*)(bp + 8 * hi);
    bv[t].q[1] = *(const uint4*)(bp + 16 + 8 * hi);
  }
  v8f o00 = {}, o01 = {}, o10 = {}, o11 = {};
  o00 = wmma_bf16(ap2[0].v, bv[0].v, o00);
  o01 = wmma_bf16(ap2[0].v, bv[1].v, o01);
  o10 = wmma_bf16(ap2[1].v, bv[0].v, o10);
  o11 = wmma_bf16(ap2[1].v, bv[1].v, o11);

  v8f oo[4] = {o00, o01, o10, o11};
#pragma unroll
  for (int mt = 0; mt < 2; ++mt)
#pragma unroll
    for (int nt = 0; nt < 2; ++nt) {
      const int dd = nt * 16 + mL;
#pragma unroll
      for (int r = 0; r < 8; ++r) {
        const int ii = mt * 16 + r + 8 * hi;
        attnout[((size_t)nw * KW + ii) * CD + h * DD + dd] =
            (__bf16)oo[mt * 2 + nt][r];
      }
    }
}

// ---------------------------------------------------------------------------
// Kernel 3: output projection. out(262144 x 256) = A_bf16 @ W_proj + b_proj
// Grid: x = n-tiles (fastest, 8) for L2 reuse of A rows; y = m-tiles.
// ---------------------------------------------------------------------------
__global__ __launch_bounds__(32) void proj_gemm_kernel(
    const __bf16* __restrict__ A, const __bf16* __restrict__ WT,
    const float* __restrict__ b_proj, float* __restrict__ out)
{
  const int lane = threadIdx.x;
  const int mL = lane & 15, hi = lane >> 4;
  const int n0 = blockIdx.x * 32;
  const int m0 = blockIdx.y * 32;

  v8f acc00 = {}, acc01 = {}, acc10 = {}, acc11 = {};

  for (int kk = 0; kk < CD; kk += 32) {
    BF16Frag a[2], b[2];
#pragma unroll
    for (int t = 0; t < 2; ++t) {
      const __bf16* ap = A + (size_t)(m0 + t * 16 + mL) * CD + kk;
      a[t].q[0] = *(const uint4*)(ap + 8 * hi);
      a[t].q[1] = *(const uint4*)(ap + 16 + 8 * hi);
      const __bf16* bp = WT + (size_t)(n0 + t * 16 + mL) * CD + kk;
      b[t].q[0] = *(const uint4*)(bp + 8 * hi);
      b[t].q[1] = *(const uint4*)(bp + 16 + 8 * hi);
    }
    acc00 = wmma_bf16(a[0].v, b[0].v, acc00);
    acc01 = wmma_bf16(a[0].v, b[1].v, acc01);
    acc10 = wmma_bf16(a[1].v, b[0].v, acc10);
    acc11 = wmma_bf16(a[1].v, b[1].v, acc11);
  }

  v8f accs[4] = {acc00, acc01, acc10, acc11};
#pragma unroll
  for (int mt = 0; mt < 2; ++mt)
#pragma unroll
    for (int nt = 0; nt < 2; ++nt) {
      const int n = n0 + nt * 16 + mL;
      const float bias = b_proj[n];
#pragma unroll
      for (int r = 0; r < 8; ++r) {
        const int m = m0 + mt * 16 + r + 8 * hi;
        out[(size_t)m * CD + n] = accs[mt * 2 + nt][r] + bias;
      }
    }
}

// ---------------------------------------------------------------------------
extern "C" void kernel_launch(void* const* d_in, const int* in_sizes, int n_in,
                              void* d_out, int out_size, void* d_ws, size_t ws_size,
                              hipStream_t stream) {
  (void)in_sizes; (void)n_in; (void)out_size; (void)ws_size;
  const float* data      = (const float*)d_in[0];
  const float* mask      = (const float*)d_in[1];
  const int*   rel_pos   = (const int*)d_in[2];
  const float* W_qkv     = (const float*)d_in[3];
  const float* b_qkv     = (const float*)d_in[4];
  const float* rpe_table = (const float*)d_in[5];
  const float* W_proj    = (const float*)d_in[6];
  const float* b_proj    = (const float*)d_in[7];
  float* out = (float*)d_out;

  char* ws = (char*)d_ws;
  size_t off = 0;
  auto wsalloc = [&](size_t bytes) -> void* {
    void* p = ws + off;
    off += (bytes + 255) & ~(size_t)255;
    return p;
  };
  __bf16* WTq     = (__bf16*)wsalloc((size_t)(3 * CD) * CD * 2);
  __bf16* WTp     = (__bf16*)wsalloc((size_t)CD * CD * 2);
  __bf16* qws     = (__bf16*)wsalloc((size_t)NW_ * HH * KW * DD * 2);
  __bf16* kws     = (__bf16*)wsalloc((size_t)NW_ * HH * KW * DD * 2);
  __bf16* vtws    = (__bf16*)wsalloc((size_t)NW_ * HH * DD * KW * 2);
  __bf16* attnout = (__bf16*)wsalloc((size_t)NW_ * KW * CD * 2);

  // Pre-transpose weights to bf16 (B^T row-major for contiguous B-frags)
  wtrans_kernel<<<(CD * 3 * CD + 255) / 256, 256, 0, stream>>>(W_qkv, WTq, CD, 3 * CD);
  wtrans_kernel<<<(CD * CD + 255) / 256, 256, 0, stream>>>(W_proj, WTp, CD, CD);

  // QKV projection GEMM (n-tiles fastest for L2 reuse)
  dim3 g1((3 * CD) / 32, NW_ * KW / 32);   // (24, 8192)
  qkv_gemm_kernel<<<g1, 32, 0, stream>>>(data, WTq, b_qkv, qws, kws, vtws);

  // Windowed attention
  dim3 g2(NW_, HH);                         // (8192, 8)
  attn_kernel<<<g2, 32, 0, stream>>>(qws, kws, vtws, mask, rel_pos, rpe_table, attnout);

  // Output projection GEMM (n-tiles fastest)
  dim3 g3(CD / 32, NW_ * KW / 32);          // (8, 8192)
  proj_gemm_kernel<<<g3, 32, 0, stream>>>(attnout, WTp, b_proj, out);
}
